// GAT_13838384627836
// MI455X (gfx1250) — matrix operand order repaired and checked by reading, not compiled, not run on hardware
//
#include <hip/hip_runtime.h>
#include <hip/hip_bf16.h>
#include <stdint.h>

#define NN_NODES 50000
#define NE_EDGES 1600000
#define NG_GRAPHS 64

typedef __attribute__((ext_vector_type(16))) _Float16 v16h;
typedef __attribute__((ext_vector_type(8)))  float    v8f;
typedef __attribute__((ext_vector_type(4)))  float    v4f;

// ---------- monotonic float<->uint encoding for atomic max ----------
__device__ __forceinline__ unsigned f2ord(float f) {
    unsigned u = __float_as_uint(f);
    return (u & 0x80000000u) ? ~u : (u | 0x80000000u);
}
__device__ __forceinline__ float ord2f(unsigned u) {
    return (u & 0x80000000u) ? __uint_as_float(u & 0x7FFFFFFFu) : __uint_as_float(~u);
}
#define ORD_NEG_INF 0x007FFFFFu  // f2ord(-inf)

// ---------- generic u32 fill (used for zeroing and -inf init) ----------
__global__ void fill_u32(unsigned* __restrict__ p, unsigned v, int n) {
    int i = blockIdx.x * blockDim.x + threadIdx.x;
    if (i < n) p[i] = v;
}

// ---------- pre-swizzle W into the WMMA B-matrix per-lane f16 layout ----------
// B 32x16 f16 layout: lane L -> col n=L%16, half=L/16 ; element j<8 -> K = half*8 + j ; j>=8 -> K = 16 + half*8 + (j-8)
// out index = ((kt*ntiles + nt)*32 + lane)*16 + j   (32B contiguous per lane)
__global__ void swizzle_w(const float* __restrict__ W, _Float16* __restrict__ out,
                          int ktiles, int ntiles, int ncols) {
    int idx = blockIdx.x * blockDim.x + threadIdx.x;
    int total = ktiles * ntiles * 32 * 16;
    if (idx >= total) return;
    int j    = idx & 15;
    int lane = (idx >> 4) & 31;
    int tn   = idx >> 9;
    int nt   = tn % ntiles;
    int kt   = tn / ntiles;
    int half = lane >> 4;
    int n    = lane & 15;
    int k    = kt * 32 + (j >> 3) * 16 + half * 8 + (j & 7);
    out[idx] = (_Float16)W[(size_t)k * ncols + nt * 16 + n];
}

// ---------- WMMA GEMM: Y[rows, NTILES*16] = X[rows, KTILES*32] @ Wsw ----------
// One wave computes a 16 x (NTILES*16) tile. rows = nRowTiles*16 (exact).
template<int KTILES, int NTILES>
__global__ void gemm_wmma(const float* __restrict__ X, const _Float16* __restrict__ Wsw,
                          float* __restrict__ Y, int nRowTiles) {
    const int K    = KTILES * 32;
    const int Ncol = NTILES * 16;
    int wave = threadIdx.x >> 5;
    int lane = threadIdx.x & 31;
    int tile = blockIdx.x * (blockDim.x >> 5) + wave;
    if (tile >= nRowTiles) return;   // whole-wave exit; no barriers in this kernel
    int half = lane >> 4;
    int l16  = lane & 15;
    const float* xrow = X + (size_t)(tile * 16 + l16) * K;   // A: M = lane%16

    v8f acc[NTILES];
#pragma unroll
    for (int nt = 0; nt < NTILES; ++nt) acc[nt] = {};

#pragma unroll
    for (int kt = 0; kt < KTILES; ++kt) {
        int kbase = kt * 32;
        if (kt + 2 < KTILES)
            __builtin_prefetch(xrow + kbase + 64, 0, 0);   // global_prefetch_b8
        // A 16x32 f16 layout: j<8 -> K=half*8+j ; j>=8 -> K=16+half*8+(j-8)
        v4f a0 = *(const v4f*)(xrow + kbase + half * 8);
        v4f a1 = *(const v4f*)(xrow + kbase + half * 8 + 4);
        v4f a2 = *(const v4f*)(xrow + kbase + 16 + half * 8);
        v4f a3 = *(const v4f*)(xrow + kbase + 16 + half * 8 + 4);
        v16h A;
        A[0]  = (_Float16)a0[0]; A[1]  = (_Float16)a0[1];
        A[2]  = (_Float16)a0[2]; A[3]  = (_Float16)a0[3];
        A[4]  = (_Float16)a1[0]; A[5]  = (_Float16)a1[1];
        A[6]  = (_Float16)a1[2]; A[7]  = (_Float16)a1[3];
        A[8]  = (_Float16)a2[0]; A[9]  = (_Float16)a2[1];
        A[10] = (_Float16)a2[2]; A[11] = (_Float16)a2[3];
        A[12] = (_Float16)a3[0]; A[13] = (_Float16)a3[1];
        A[14] = (_Float16)a3[2]; A[15] = (_Float16)a3[3];
#pragma unroll
        for (int nt = 0; nt < NTILES; ++nt) {
            v16h Bv = *(const v16h*)(Wsw + ((size_t)(kt * NTILES + nt) * 32 + lane) * 16);
            acc[nt] = __builtin_amdgcn_wmma_f32_16x16x32_f16(
                false, A, false, Bv, (short)0, acc[nt], false, false);
        }
    }
    // C/D layout: element v of lane -> M = v + 8*half, N = lane%16
#pragma unroll
    for (int nt = 0; nt < NTILES; ++nt) {
#pragma unroll
        for (int v = 0; v < 8; ++v) {
            int r = tile * 16 + half * 8 + v;
            Y[(size_t)r * Ncol + nt * 16 + l16] = acc[nt][v];
        }
    }
}

// ---------- per-(node,head) attention scores ----------
template<int H, int C>
__global__ void attn_k(const float* __restrict__ feat, const float* __restrict__ a_src,
                       const float* __restrict__ a_dst, float* __restrict__ as,
                       float* __restrict__ ad, int Nn) {
    int i = blockIdx.x * blockDim.x + threadIdx.x;
    if (i >= Nn * H) return;
    int h = i % H;
    const float* fp = feat + (size_t)(i / H) * (H * C) + h * C;
    float s = 0.f, d = 0.f;
#pragma unroll
    for (int c = 0; c < C; ++c) {
        float v = fp[c];
        s += v * a_src[h * C + c];
        d += v * a_dst[h * C + c];
    }
    as[i] = s;
    ad[i] = d;
}

// ---------- pass 1: segment max over dst (self-loops appended) ----------
template<int H>
__global__ void edge_max_k(const int* __restrict__ src, const int* __restrict__ dst,
                           const float* __restrict__ as, const float* __restrict__ ad,
                           unsigned* __restrict__ emax, int E, int Nn) {
    int i = blockIdx.x * blockDim.x + threadIdx.x;
    if (i >= (E + Nn) * H) return;
    int e = i / H, h = i % H;
    int s, d;
    if (e < E) { s = src[e]; d = dst[e]; } else { s = d = e - E; }
    float v = as[s * H + h] + ad[d * H + h];
    v = v > 0.f ? v : 0.2f * v;            // leaky_relu(0.2)
    atomicMax(&emax[d * H + h], f2ord(v));
}

// ---------- pass 2: exp-weighted gather + scatter-add (unnormalized) ----------
template<int H, int C>
__global__ void edge_agg_k(const int* __restrict__ src, const int* __restrict__ dst,
                           const float* __restrict__ as, const float* __restrict__ ad,
                           const unsigned* __restrict__ emax, const float* __restrict__ feat,
                           float* __restrict__ denom, float* __restrict__ acc, int E, int Nn) {
    int i = blockIdx.x * blockDim.x + threadIdx.x;
    if (i >= (E + Nn) * H) return;
    int e = i / H, h = i % H;
    int s, d;
    if (e < E) { s = src[e]; d = dst[e]; } else { s = d = e - E; }
    float v = as[s * H + h] + ad[d * H + h];
    v = v > 0.f ? v : 0.2f * v;
    float w = __expf(v - ord2f(emax[d * H + h]));
    atomicAdd(&denom[d * H + h], w);
    const float* fp = feat + (size_t)s * (H * C) + h * C;
    float* ap = acc + (size_t)d * (H * C) + h * C;
#pragma unroll
    for (int c = 0; c < C; ++c) atomicAdd(ap + c, fp[c] * w);
}

// ---------- layer1 epilogue: normalize + bias + ELU (in place) ----------
__global__ void finalize1_k(float* __restrict__ acc, const float* __restrict__ denom,
                            const float* __restrict__ bias, int Nn) {
    int i = blockIdx.x * blockDim.x + threadIdx.x;
    if (i >= Nn * 64) return;
    int c = i & 63, node = i >> 6, h = c >> 3;
    float v = acc[i] / (denom[node * 8 + h] + 1e-16f) + bias[c];
    acc[i] = v > 0.f ? v : (__expf(v) - 1.f);   // ELU(alpha=1)
}

// ---------- layer2 epilogue: normalize + bias, then graph-sum pool ----------
__global__ void finalize2_pool_k(const float* __restrict__ acc, const float* __restrict__ denom,
                                 const float* __restrict__ bias, const int* __restrict__ batch,
                                 float* __restrict__ pool, int Nn) {
    int i = blockIdx.x * blockDim.x + threadIdx.x;
    if (i >= Nn * 32) return;
    int c = i & 31, node = i >> 5;
    float v = acc[i] / (denom[node] + 1e-16f) + bias[c];
    atomicAdd(&pool[batch[node] * 32 + c], v);
}

__global__ void count_k(const int* __restrict__ batch, float* __restrict__ cnt, int Nn) {
    int i = blockIdx.x * blockDim.x + threadIdx.x;
    if (i < Nn) atomicAdd(&cnt[batch[i]], 1.0f);
}

// ---------- final linear head: out[64,2] ----------
__global__ void head_k(const float* __restrict__ pool, const float* __restrict__ cnt,
                       const float* __restrict__ lw, const float* __restrict__ lb,
                       float* __restrict__ out) {
    int i = threadIdx.x;
    if (i >= NG_GRAPHS * 2) return;
    int g = i >> 1, j = i & 1;
    float c = cnt[g]; c = c > 1.f ? c : 1.f;
    float s = 0.f;
#pragma unroll
    for (int k = 0; k < 32; ++k) s += (pool[g * 32 + k] / c) * lw[k * 2 + j];
    out[i] = s + lb[j];
}

extern "C" void kernel_launch(void* const* d_in, const int* in_sizes, int n_in,
                              void* d_out, int out_size, void* d_ws, size_t ws_size,
                              hipStream_t stream) {
    const float* x     = (const float*)d_in[0];
    const int*   eidx  = (const int*)d_in[1];
    const int*   batch = (const int*)d_in[2];
    const float* W1    = (const float*)d_in[3];
    const float* asrc1 = (const float*)d_in[4];
    const float* adst1 = (const float*)d_in[5];
    const float* bias1 = (const float*)d_in[6];
    const float* W2    = (const float*)d_in[7];
    const float* asrc2 = (const float*)d_in[8];
    const float* adst2 = (const float*)d_in[9];
    const float* bias2 = (const float*)d_in[10];
    const float* linw  = (const float*)d_in[11];
    const float* linb  = (const float*)d_in[12];
    float* out = (float*)d_out;

    const int NN = NN_NODES, E = NE_EDGES;
    const int* src = eidx;
    const int* dst = eidx + E;

    // ---- workspace carve (256B aligned bump allocator) ----
    char* p = (char*)d_ws;
    auto carve = [&](size_t bytes) -> char* {
        char* r = p;
        p += (bytes + 255) & ~(size_t)255;
        return r;
    };
    float*     h1     = (float*)carve((size_t)NN * 64 * 4);
    float*     as1    = (float*)carve((size_t)NN * 8 * 4);
    float*     ad1    = (float*)carve((size_t)NN * 8 * 4);
    unsigned*  emax1  = (unsigned*)carve((size_t)NN * 8 * 4);
    float*     denom1 = (float*)carve((size_t)NN * 8 * 4);
    float*     acc1   = (float*)carve((size_t)NN * 64 * 4);   // becomes h1e in place
    float*     h2     = (float*)carve((size_t)NN * 32 * 4);
    float*     as2    = (float*)carve((size_t)NN * 4);
    float*     ad2    = (float*)carve((size_t)NN * 4);
    unsigned*  emax2  = (unsigned*)carve((size_t)NN * 4);
    float*     denom2 = (float*)carve((size_t)NN * 4);
    float*     acc2   = (float*)carve((size_t)NN * 32 * 4);
    float*     pool   = (float*)carve((size_t)NG_GRAPHS * 32 * 4);
    float*     cnt    = (float*)carve((size_t)NG_GRAPHS * 4);
    _Float16*  Wsw1   = (_Float16*)carve((size_t)16 * 4 * 32 * 16 * 2);
    _Float16*  Wsw2   = (_Float16*)carve((size_t)2 * 2 * 32 * 16 * 2);

    const int B = 256;
    auto nb = [](long n) { return (int)((n + 255) / 256); };

    // ---- re-init all accumulator state (graph replay safe) ----
    fill_u32<<<nb(NN * 8),  B, 0, stream>>>(emax1, ORD_NEG_INF, NN * 8);
    fill_u32<<<nb(NN * 8),  B, 0, stream>>>((unsigned*)denom1, 0u, NN * 8);
    fill_u32<<<nb(NN * 64), B, 0, stream>>>((unsigned*)acc1, 0u, NN * 64);
    fill_u32<<<nb(NN),      B, 0, stream>>>(emax2, ORD_NEG_INF, NN);
    fill_u32<<<nb(NN),      B, 0, stream>>>((unsigned*)denom2, 0u, NN);
    fill_u32<<<nb(NN * 32), B, 0, stream>>>((unsigned*)acc2, 0u, NN * 32);
    fill_u32<<<1, B, 0, stream>>>((unsigned*)pool, 0u, NG_GRAPHS * 32);
    fill_u32<<<1, B, 0, stream>>>((unsigned*)cnt, 0u, NG_GRAPHS);

    // ---- swizzle weights to WMMA B-operand layout (f16) ----
    swizzle_w<<<nb(16 * 4 * 32 * 16), B, 0, stream>>>(W1, Wsw1, 16, 4, 64);
    swizzle_w<<<nb(2 * 2 * 32 * 16),  B, 0, stream>>>(W2, Wsw2, 2, 2, 32);

    const int rowTiles = NN / 16;                   // 3125 exactly
    const int gemmBlocks = (rowTiles + 7) / 8;      // 8 waves/block

    // ---- layer 1 ----
    gemm_wmma<16, 4><<<gemmBlocks, B, 0, stream>>>(x, Wsw1, h1, rowTiles);
    attn_k<8, 8><<<nb((long)NN * 8), B, 0, stream>>>(h1, asrc1, adst1, as1, ad1, NN);
    edge_max_k<8><<<nb((long)(E + NN) * 8), B, 0, stream>>>(src, dst, as1, ad1, emax1, E, NN);
    edge_agg_k<8, 8><<<nb((long)(E + NN) * 8), B, 0, stream>>>(src, dst, as1, ad1, emax1, h1,
                                                               denom1, acc1, E, NN);
    finalize1_k<<<nb((long)NN * 64), B, 0, stream>>>(acc1, denom1, bias1, NN);

    // ---- layer 2 ----
    gemm_wmma<2, 2><<<gemmBlocks, B, 0, stream>>>(acc1, Wsw2, h2, rowTiles);
    attn_k<1, 32><<<nb(NN), B, 0, stream>>>(h2, asrc2, adst2, as2, ad2, NN);
    edge_max_k<1><<<nb(E + NN), B, 0, stream>>>(src, dst, as2, ad2, emax2, E, NN);
    edge_agg_k<1, 32><<<nb(E + NN), B, 0, stream>>>(src, dst, as2, ad2, emax2, h2,
                                                    denom2, acc2, E, NN);

    // ---- pool + head ----
    count_k<<<nb(NN), B, 0, stream>>>(batch, cnt, NN);
    finalize2_pool_k<<<nb((long)NN * 32), B, 0, stream>>>(acc2, denom2, bias2, batch, pool, NN);
    head_k<<<1, 128, 0, stream>>>(pool, cnt, linw, linb, out);
}
